// HGAT_40819369181431
// MI455X (gfx1250) — matrix operand order repaired and checked by reading, not compiled, not run on hardware
//
#include <hip/hip_runtime.h>
#include <hip/hip_bf16.h>

typedef __attribute__((ext_vector_type(16))) _Float16 v16h;
typedef __attribute__((ext_vector_type(8)))  float    v8f;

#define NN      50000
#define SEQ     64
#define NEDGE   2000
#define NINC    800000
#define CH      32

static __device__ __forceinline__ v8f wmma_f16(v16h a, v16h b, v8f c) {
    return __builtin_amdgcn_wmma_f32_16x16x32_f16(false, a, false, b, (short)0, c, false, false);
}

// ---------------------------------------------------------------------------
// Pack a row-major weight matrix into WMMA B-matrix lane layout (f16).
// B tile jj computes (X @ W^T) columns [ (jj%tpg)*16 , +16 ) with
// K offset (jj/tpg)*colStride.  Lane l holds column n=l&15; lane half g carries
// K=8g..8g+7 and 16+8g..16+8g+7, two halves per dword p (K=16*(p>>2)+8g+2*(p&3)+h).
// ---------------------------------------------------------------------------
__global__ void pack_b_kernel(const float* __restrict__ W, int ld, int tpg,
                              int colStride, _Float16* __restrict__ out, int ntiles) {
    int idx = blockIdx.x * blockDim.x + threadIdx.x;          // ntiles*32*8
    if (idx >= ntiles * 256) return;
    int p  = idx & 7;
    int l  = (idx >> 3) & 31;
    int jj = idx >> 8;
    int nl = l & 15, g = l >> 4;
    int row     = (jj % tpg) * 16 + nl;
    int colBase = (jj / tpg) * colStride;
    int k0      = ((p >> 2) << 4) + (g << 3) + ((p & 3) << 1);
    out[idx * 2 + 0] = (_Float16)W[row * ld + colBase + k0];
    out[idx * 2 + 1] = (_Float16)W[row * ld + colBase + k0 + 1];
}

// ---------------------------------------------------------------------------
// Fused GRU + lin2: one wave per 16-node tile. Per step:
//   6 WMMAs  H(t-1)[16x32] x Whh^T tiles  -> gates [16x96]
//   scalar   x_t @ Wih^T (K=6)            -> gi
//   gate math, H(t) transposed through LDS back to A layout,
//   2 WMMAs  H(t) x lin2B(t)              -> OUT accumulation
// ---------------------------------------------------------------------------
__global__ __launch_bounds__(256) void gru_lin2_kernel(
    const float* __restrict__ price,      // [N,64,6]
    const float* __restrict__ W_ih,       // [96,6]
    const float* __restrict__ b_ih,
    const float* __restrict__ b_hh,
    const v16h*  __restrict__ whhB,       // [6][32] packed
    const v16h*  __restrict__ lin2B,      // [128][32] packed (64 steps x 2 tiles)
    const float* __restrict__ lin2_b,
    float* __restrict__ feat)             // [N,32]
{
    __shared__ float xsh[8][16 * 6];
    __shared__ float hsh[8][16 * CH];

    const int wave = threadIdx.x >> 5;
    const int lane = threadIdx.x & 31;
    const int g    = lane >> 4;
    const int nl   = lane & 15;
    const int nodeBase = blockIdx.x * 128 + wave * 16;

    // per-lane gate rows: for tc in {0,1}: c = 16*tc+nl ; gates c, 32+c, 64+c
    float wih[2][3][6], bi[2][3], bh[2][3];
    for (int tc = 0; tc < 2; ++tc) {
        int c = 16 * tc + nl;
        for (int q = 0; q < 3; ++q) {
            int gr = 32 * q + c;
            for (int k = 0; k < 6; ++k) wih[tc][q][k] = W_ih[gr * 6 + k];
            bi[tc][q] = b_ih[gr];
            bh[tc][q] = b_hh[gr];
        }
    }

    v16h Bh[6];
    for (int j = 0; j < 6; ++j) Bh[j] = whhB[(j << 5) + lane];

    v16h A;
    for (int i = 0; i < 16; ++i) A[i] = (_Float16)0.0f;
    float hC[2][8];
    for (int tc = 0; tc < 2; ++tc)
        for (int r = 0; r < 8; ++r) hC[tc][r] = 0.0f;
    v8f out0 = {0,0,0,0,0,0,0,0}, out1 = {0,0,0,0,0,0,0,0};
    v8f zero = {0,0,0,0,0,0,0,0};

    for (int t = 0; t < SEQ; ++t) {
        // stage x_t for 16 nodes into LDS (clamped OOB loads, stores guarded later)
        if (lane < 16) {
            int nd = nodeBase + lane; if (nd >= NN) nd = NN - 1;
            const float* xp = price + ((size_t)nd * SEQ + t) * 6;
            float2 a0 = *(const float2*)(xp);
            float2 a1 = *(const float2*)(xp + 2);
            float2 a2 = *(const float2*)(xp + 4);
            xsh[wave][lane * 6 + 0] = a0.x; xsh[wave][lane * 6 + 1] = a0.y;
            xsh[wave][lane * 6 + 2] = a1.x; xsh[wave][lane * 6 + 3] = a1.y;
            xsh[wave][lane * 6 + 4] = a2.x; xsh[wave][lane * 6 + 5] = a2.y;
        }
        asm volatile("s_wait_dscnt 0" ::: "memory");   // wave-local LDS RAW

        // recurrent gates: gh = H(t-1) @ Whh^T  (6 C tiles)
        v8f Cg[6];
        for (int j = 0; j < 6; ++j) Cg[j] = wmma_f16(A, Bh[j], zero);

        // input gates + nonlinearity + new hidden
        for (int r = 0; r < 8; ++r) {
            int m = 8 * g + r;
            float x0 = xsh[wave][m * 6 + 0], x1 = xsh[wave][m * 6 + 1];
            float x2 = xsh[wave][m * 6 + 2], x3 = xsh[wave][m * 6 + 3];
            float x4 = xsh[wave][m * 6 + 4], x5 = xsh[wave][m * 6 + 5];
            for (int tc = 0; tc < 2; ++tc) {
                float gir = bi[tc][0] + x0*wih[tc][0][0] + x1*wih[tc][0][1] + x2*wih[tc][0][2]
                                      + x3*wih[tc][0][3] + x4*wih[tc][0][4] + x5*wih[tc][0][5];
                float giz = bi[tc][1] + x0*wih[tc][1][0] + x1*wih[tc][1][1] + x2*wih[tc][1][2]
                                      + x3*wih[tc][1][3] + x4*wih[tc][1][4] + x5*wih[tc][1][5];
                float gin = bi[tc][2] + x0*wih[tc][2][0] + x1*wih[tc][2][1] + x2*wih[tc][2][2]
                                      + x3*wih[tc][2][3] + x4*wih[tc][2][4] + x5*wih[tc][2][5];
                float ghr = Cg[0 + tc][r] + bh[tc][0];
                float ghz = Cg[2 + tc][r] + bh[tc][1];
                float ghn = Cg[4 + tc][r] + bh[tc][2];
                float rg = 1.0f / (1.0f + __expf(-(gir + ghr)));
                float zg = 1.0f / (1.0f + __expf(-(giz + ghz)));
                float ng = tanhf(gin + rg * ghn);
                float hn = (1.0f - zg) * ng + zg * hC[tc][r];
                hC[tc][r] = hn;
                hsh[wave][m * CH + 16 * tc + nl] = hn;
            }
        }
        asm volatile("s_wait_dscnt 0" ::: "memory");   // wave-local LDS RAW

        // transpose H(t) back into A-matrix lane layout
        v16h An;
        for (int p = 0; p < 8; ++p) {
            int k0 = ((p >> 2) << 4) + (g << 3) + ((p & 3) << 1);
            An[2 * p + 0] = (_Float16)hsh[wave][nl * CH + k0];
            An[2 * p + 1] = (_Float16)hsh[wave][nl * CH + k0 + 1];
        }
        A = An;

        // fused lin2 accumulation: OUT += H(t) @ lin2_w[:, 32t:32t+32]^T
        out0 = wmma_f16(A, lin2B[(2 * t + 0) * 32 + lane], out0);
        out1 = wmma_f16(A, lin2B[(2 * t + 1) * 32 + lane], out1);
    }

    // epilogue: bias + leaky_relu(0.01) -> feat
    for (int tc = 0; tc < 2; ++tc) {
        int col = 16 * tc + nl;
        float bb = lin2_b[col];
        for (int r = 0; r < 8; ++r) {
            int node = nodeBase + 8 * g + r;
            float v = (tc ? out1[r] : out0[r]) + bb;
            v = v > 0.0f ? v : 0.01f * v;
            if (node < NN) feat[(size_t)node * CH + col] = v;
        }
    }
}

// ---------------------------------------------------------------------------
// Generic [n,32] x [32,32]^T via WMMA, optional bias + leaky relu.
// ---------------------------------------------------------------------------
__global__ __launch_bounds__(256) void linear_wmma_kernel(
    const float* __restrict__ in, const v16h* __restrict__ Bt,   // [2][32] packed
    const float* __restrict__ bias, float slope, int act,
    float* __restrict__ out, int n)
{
    const int wave = threadIdx.x >> 5;
    const int lane = threadIdx.x & 31;
    const int g = lane >> 4, nl = lane & 15;
    const int nodeBase = blockIdx.x * 128 + wave * 16;

    int myNode = nodeBase + nl; if (myNode >= n) myNode = n - 1;
    const float* row = in + (size_t)myNode * CH;
    v16h A;
    for (int p = 0; p < 8; ++p) {
        int k0 = ((p >> 2) << 4) + (g << 3) + ((p & 3) << 1);
        float2 f = *(const float2*)(row + k0);
        A[2 * p + 0] = (_Float16)f.x;
        A[2 * p + 1] = (_Float16)f.y;
    }
    v8f zero = {0,0,0,0,0,0,0,0};
    v8f c0 = wmma_f16(A, Bt[lane], zero);
    v8f c1 = wmma_f16(A, Bt[32 + lane], zero);

    for (int tc = 0; tc < 2; ++tc) {
        int col = 16 * tc + nl;
        float bb = bias ? bias[col] : 0.0f;
        for (int r = 0; r < 8; ++r) {
            int node = nodeBase + 8 * g + r;
            float v = (tc ? c1[r] : c0[r]) + bb;
            if (act) v = v > 0.0f ? v : slope * v;
            if (node < n) out[(size_t)node * CH + col] = v;
        }
    }
}

// ---------------------------------------------------------------------------
// Hypergraph helpers
// ---------------------------------------------------------------------------
__global__ void degree_kernel(const int* __restrict__ ni, const int* __restrict__ ei,
                              int* dcnt, int* bcnt, int E) {
    int i = blockIdx.x * blockDim.x + threadIdx.x;
    if (i < E) {
        atomicAdd(&dcnt[ni[i]], 1);
        atomicAdd(&bcnt[ei[i]], 1);
    }
}

__global__ void invert_kernel(const int* __restrict__ cnt, float* inv, int n) {
    int i = blockIdx.x * blockDim.x + threadIdx.x;
    if (i < n) {
        int c = cnt[i];
        inv[i] = c > 0 ? 1.0f / (float)c : 0.0f;
    }
}

__global__ void bounds_kernel(const int* __restrict__ ei, int* es, int* ee, int E) {
    int i = blockIdx.x * blockDim.x + threadIdx.x;
    if (i < E) {
        int e = ei[i];
        if (i == 0     || ei[i - 1] != e) es[e] = i;
        if (i == E - 1 || ei[i + 1] != e) ee[e] = i + 1;
    }
}

// one block per hyperedge; edge_idx is sorted so [es,ee) is contiguous
__global__ __launch_bounds__(256) void edge_agg_kernel(
    const float* __restrict__ xw, const int* __restrict__ ni,
    const int* __restrict__ es, const int* __restrict__ ee,
    const float* __restrict__ binv, float* __restrict__ edge_out)
{
    __shared__ float red[8][CH];
    int e = blockIdx.x;
    int c = threadIdx.x & 31, w = threadIdx.x >> 5;
    int s = es[e], t = ee[e];
    float acc = 0.0f;
    for (int i = s + w; i < t; i += 8)
        acc += xw[(size_t)ni[i] * CH + c];
    red[w][c] = acc;
    __syncthreads();
    if (w == 0) {
        float s2 = 0.0f;
        for (int q = 0; q < 8; ++q) s2 += red[q][c];
        edge_out[e * CH + c] = s2 * binv[e];
    }
}

__global__ void scatter_kernel(const float* __restrict__ edge_out,
                               const int* __restrict__ ni, const int* __restrict__ ei,
                               float* acc, int E) {
    int idx = blockIdx.x * blockDim.x + threadIdx.x;
    int i = idx >> 5, c = idx & 31;
    if (i < E)
        atomicAdd(&acc[(size_t)ni[i] * CH + c], edge_out[ei[i] * CH + c]);
}

__global__ void finalize_kernel(const float* __restrict__ acc, const float* __restrict__ dinv,
                                const float* __restrict__ bias, float* out, int n) {
    int idx = blockIdx.x * blockDim.x + threadIdx.x;
    if (idx < n * CH) {
        int node = idx >> 5, c = idx & 31;
        float v = acc[idx] * dinv[node] + bias[c];
        out[idx] = v > 0.0f ? v : 0.2f * v;
    }
}

// ---------------------------------------------------------------------------
extern "C" void kernel_launch(void* const* d_in, const int* in_sizes, int n_in,
                              void* d_out, int out_size, void* d_ws, size_t ws_size,
                              hipStream_t stream) {
    const float* price  = (const float*)d_in[0];
    const int*   e      = (const int*)  d_in[1];   // [2, NINC]
    // d_in[2] = concept (unused by reference)
    const float* W_ih   = (const float*)d_in[3];
    const float* W_hh   = (const float*)d_in[4];
    const float* b_ih   = (const float*)d_in[5];
    const float* b_hh   = (const float*)d_in[6];
    const float* lin2_w = (const float*)d_in[7];
    const float* lin2_b = (const float*)d_in[8];
    const float* h1_w   = (const float*)d_in[9];
    const float* h1_b   = (const float*)d_in[10];
    const float* h2_w   = (const float*)d_in[11];
    const float* h2_b   = (const float*)d_in[12];
    const float* lin1_w = (const float*)d_in[13];
    const float* lin1_b = (const float*)d_in[14];
    float* outp = (float*)d_out;

    const int* node_idx = e;
    const int* edge_idx = e + NINC;

    // workspace carve-up (256B aligned)
    char* ws = (char*)d_ws;
    size_t off = 0;
    auto carve = [&](size_t bytes) -> void* {
        void* p = ws + off;
        off = (off + bytes + 255) & ~(size_t)255;
        return p;
    };
    _Float16* lin2B = (_Float16*)carve(128 * 32 * 16 * sizeof(_Float16));
    _Float16* whhB  = (_Float16*)carve(6   * 32 * 16 * sizeof(_Float16));
    _Float16* h1B   = (_Float16*)carve(2   * 32 * 16 * sizeof(_Float16));
    _Float16* h2B   = (_Float16*)carve(2   * 32 * 16 * sizeof(_Float16));
    _Float16* lin1B = (_Float16*)carve(2   * 32 * 16 * sizeof(_Float16));
    float* feat = (float*)carve((size_t)NN * CH * 4);
    float* xw   = (float*)carve((size_t)NN * CH * 4);
    float* acc  = (float*)carve((size_t)NN * CH * 4);
    float* eout = (float*)carve((size_t)NEDGE * CH * 4);
    int*   dcnt = (int*)  carve((size_t)NN * 4);
    int*   bcnt = (int*)  carve((size_t)NEDGE * 4);
    float* dinv = (float*)carve((size_t)NN * 4);
    float* binv = (float*)carve((size_t)NEDGE * 4);
    int*   es   = (int*)  carve((size_t)NEDGE * 4);
    int*   ee   = (int*)  carve((size_t)NEDGE * 4);

    // --- weight packing into WMMA B layout -------------------------------
    pack_b_kernel<<<(128 * 256 + 255) / 256, 256, 0, stream>>>(lin2_w, 2048, 2, 32, lin2B, 128);
    pack_b_kernel<<<(6   * 256 + 255) / 256, 256, 0, stream>>>(W_hh,   32,   6, 0,  whhB,  6);
    pack_b_kernel<<<2, 256, 0, stream>>>(h1_w,   32, 2, 0, h1B,   2);
    pack_b_kernel<<<2, 256, 0, stream>>>(h2_w,   32, 2, 0, h2B,   2);
    pack_b_kernel<<<2, 256, 0, stream>>>(lin1_w, 32, 2, 0, lin1B, 2);

    // --- graph structure --------------------------------------------------
    hipMemsetAsync(dcnt, 0, (size_t)NN * 4, stream);
    hipMemsetAsync(bcnt, 0, (size_t)NEDGE * 4, stream);
    hipMemsetAsync(es,   0, (size_t)NEDGE * 4, stream);
    hipMemsetAsync(ee,   0, (size_t)NEDGE * 4, stream);
    degree_kernel<<<(NINC + 255) / 256, 256, 0, stream>>>(node_idx, edge_idx, dcnt, bcnt, NINC);
    invert_kernel<<<(NN + 255) / 256, 256, 0, stream>>>(dcnt, dinv, NN);
    invert_kernel<<<(NEDGE + 255) / 256, 256, 0, stream>>>(bcnt, binv, NEDGE);
    bounds_kernel<<<(NINC + 255) / 256, 256, 0, stream>>>(edge_idx, es, ee, NINC);

    // --- fused GRU + lin2 -------------------------------------------------
    const int gblocks = (NN + 127) / 128;
    gru_lin2_kernel<<<gblocks, 256, 0, stream>>>(price, W_ih, b_ih, b_hh,
                                                 (const v16h*)whhB, (const v16h*)lin2B,
                                                 lin2_b, feat);

    // --- hypergraph conv 1 ------------------------------------------------
    linear_wmma_kernel<<<gblocks, 256, 0, stream>>>(feat, (const v16h*)h1B, nullptr, 0.f, 0, xw, NN);
    hipMemsetAsync(acc, 0, (size_t)NN * CH * 4, stream);
    edge_agg_kernel<<<NEDGE, 256, 0, stream>>>(xw, node_idx, es, ee, binv, eout);
    scatter_kernel<<<(NINC * 32 + 255) / 256, 256, 0, stream>>>(eout, node_idx, edge_idx, acc, NINC);
    finalize_kernel<<<(NN * CH + 255) / 256, 256, 0, stream>>>(acc, dinv, h1_b, feat, NN);

    // --- hypergraph conv 2 ------------------------------------------------
    linear_wmma_kernel<<<gblocks, 256, 0, stream>>>(feat, (const v16h*)h2B, nullptr, 0.f, 0, xw, NN);
    hipMemsetAsync(acc, 0, (size_t)NN * CH * 4, stream);
    edge_agg_kernel<<<NEDGE, 256, 0, stream>>>(xw, node_idx, es, ee, binv, eout);
    scatter_kernel<<<(NINC * 32 + 255) / 256, 256, 0, stream>>>(eout, node_idx, edge_idx, acc, NINC);
    finalize_kernel<<<(NN * CH + 255) / 256, 256, 0, stream>>>(acc, dinv, h2_b, feat, NN);

    // --- final linear + leaky(0.01) -> d_out ------------------------------
    linear_wmma_kernel<<<gblocks, 256, 0, stream>>>(feat, (const v16h*)lin1B, lin1_b, 0.01f, 1, outp, NN);
}
